// LSTMNetwork_65206193487960
// MI455X (gfx1250) — matrix operand order repaired and checked by reading, not compiled, not run on hardware
//
#include <hip/hip_runtime.h>
#include <hip/hip_bf16.h>

// ---------------------------------------------------------------------------
// LSTM on MI455X (gfx1250): bf16 WMMA GEMMs + TDM (tensor_load_to_lds) weight
// staging. Per-timestep fused kernel: gates = x_t@W_ih^T + h@W_hh^T + b, cell
// update, done-masking folded into state handoff. Double-buffered h/c state.
// Each block: 8 waves share one h_tile -> weight slice staged once into LDS
// by the Tensor Data Mover (double-buffered k-chunks), B frags read via DS.
// ---------------------------------------------------------------------------

#define T_STEPS 256
#define BATCH   256
#define INDIM   512
#define HDIM    512
#define G4      (4 * HDIM)       // 2048
#define KCHUNK  128              // k elements staged per TDM op
#define NCHUNK  8                // 4 chunks x_t pass + 4 chunks h pass
#define LDS_CHUNK_ELEMS (4 * 16 * KCHUNK)   // 8192 bf16 = 16KB

typedef __attribute__((ext_vector_type(16))) __bf16        v16bf;
typedef __attribute__((ext_vector_type(8)))  float         v8f;
typedef __attribute__((ext_vector_type(4)))  unsigned int  v4u;
typedef __attribute__((ext_vector_type(8)))  int           v8i;
typedef __attribute__((ext_vector_type(4)))  int           v4i;

#if defined(__has_builtin)
#if __has_builtin(__builtin_amdgcn_tensor_load_to_lds)
#define USE_TDM 1
#endif
#endif

__device__ __forceinline__ unsigned short f32_to_bf16_rne(float f) {
    union { float f; unsigned int u; } x; x.f = f;
    unsigned int lsb = (x.u >> 16) & 1u;
    unsigned int r   = x.u + 0x7FFFu + lsb;
    return (unsigned short)(r >> 16);
}

__device__ __forceinline__ float fast_sigmoid(float x) {
    return 1.0f / (1.0f + __expf(-x));
}

// A-fragment (16-bit 16x32): lane<16 -> K 0..15,
// lane>=16 -> K 16..31; two contiguous 16B chunks.
__device__ __forceinline__ v16bf load_frag_a(const unsigned short* p) {
    union { uint4 q[2]; v16bf v; } u;
    u.q[0] = *reinterpret_cast<const uint4*>(p);
    u.q[1] = *reinterpret_cast<const uint4*>(p + 8);
    return u.v;
}

// B-fragment (16-bit 32x16): lane = N, lanes 0-15 hold K=0..15,
// lanes 16-31 hold K=16..31 (contiguous 32B per lane). Source: LDS.
__device__ __forceinline__ v16bf load_frag_b(const unsigned short* p) {
    union { uint4 q[2]; v16bf v; } u;
    u.q[0] = *reinterpret_cast<const uint4*>(p);
    u.q[1] = *reinterpret_cast<const uint4*>(p + 8);
    return u.v;
}

#ifdef USE_TDM
// Issue one TDM load: 3D tile (k=KCHUNK fast, 16 rows, 4 gates) from a
// row-major [2048 x 512] bf16 weight matrix into LDS at lds_byte_addr.
// D# layout per CDNA5 ISA. This toolchain exposes the
// 6-arg builtin: (v4u g0, v8i g1, v4i g2, v4i g3, v8i extra, i32 cpol).
__device__ __forceinline__ void tdm_load_w_slice(unsigned lds_byte_addr,
                                                 const unsigned short* gptr) {
    unsigned long long ga = (unsigned long long)(size_t)gptr;
    v4u g0;
    g0[0] = 1u;                                  // count=1, user descriptor
    g0[1] = lds_byte_addr;                       // lds_addr
    g0[2] = (unsigned)(ga & 0xFFFFFFFFu);        // global_addr[31:0]
    g0[3] = (unsigned)((ga >> 32) & 0x01FFFFFFu) // global_addr[56:32]
            | (2u << 30);                        // type=2 ("image")
    v8i g1;
    g1[0] = 0x00010000;                          // data_size=1 (2 bytes)
    g1[1] = (int)(512u  << 16);                  // tensor_dim0 = 512 (k)
    g1[2] = (int)(2048u << 16);                  // tensor_dim1 = 2048 (rows)
    g1[3] = (int)((unsigned)KCHUNK << 16);       // tile_dim0 = KCHUNK
    g1[4] = (int)(16u | (4u << 16));             // tile_dim1=16, tile_dim2=4
    g1[5] = 512;                                 // tensor_dim0_stride = 512
    g1[6] = 0;                                   // dim1_stride[15:0]=0 (262144)
    g1[7] = 4;                                   // dim1_stride[47:16] = 4
    v4i g2;
    g2[0] = 4;                                   // tensor_dim2 = 4 (gates)
    g2[1] = 0; g2[2] = 0; g2[3] = 0;
    v4i g3; g3[0] = 0; g3[1] = 0; g3[2] = 0; g3[3] = 0;
    v8i gx; gx[0] = 0; gx[1] = 0; gx[2] = 0; gx[3] = 0;
            gx[4] = 0; gx[5] = 0; gx[6] = 0; gx[7] = 0;
    __builtin_amdgcn_tensor_load_to_lds(g0, g1, g2, g3, gx, 0);
}
#endif

// ---------------------------------------------------------------------------
// Prologue kernels
// ---------------------------------------------------------------------------
__global__ void cvt_f32_bf16_kernel(const float* __restrict__ s,
                                    unsigned short* __restrict__ d, int n) {
    for (int i = blockIdx.x * blockDim.x + threadIdx.x; i < n;
         i += gridDim.x * blockDim.x)
        d[i] = f32_to_bf16_rne(s[i]);
}

__global__ void init_kernel(const float* __restrict__ b_ih,
                            const float* __restrict__ b_hh,
                            float* __restrict__ bias,
                            unsigned short* __restrict__ h0,
                            float* __restrict__ c0,
                            float* __restrict__ hidden0) {  // 2*B*H floats
    int i = blockIdx.x * blockDim.x + threadIdx.x;
    if (i < G4) bias[i] = b_ih[i] + b_hh[i];
    if (i < BATCH * HDIM) { h0[i] = 0; c0[i] = 0.0f; }
    if (i < 2 * BATCH * HDIM) hidden0[i] = 0.0f;
}

// ---------------------------------------------------------------------------
// One timestep. grid = 64 blocks x 256 threads (8 waves). All waves of a
// block share h_tile; weight slices for the 4 gates are TDM-staged to LDS.
// Each wave owns a 16x16 (batch x hidden) patch, 4 gate accumulators.
// ---------------------------------------------------------------------------
__global__ void __launch_bounds__(256)
lstm_step_kernel(const unsigned short* __restrict__ x_t,    // [B, IN] bf16
                 const unsigned short* __restrict__ Wih,    // [4H, IN] bf16
                 const unsigned short* __restrict__ Whh,    // [4H, H]  bf16
                 const float* __restrict__ bias,            // [4H]
                 const unsigned short* __restrict__ h_in,   // [B, H] bf16 (masked)
                 const float* __restrict__ c_in,            // [B, H] f32  (masked)
                 unsigned short* __restrict__ h_out,        // [B, H] bf16 (t+1)
                 float* __restrict__ c_out,                 // [B, H] f32
                 const int* __restrict__ dones_next,        // [B] (dones[t+1])
                 float* __restrict__ lstm_out_t,            // d_out + t*B*H
                 float* __restrict__ hidden_out_next,       // [2,B,H] at t+1
                 int has_next) {
    __shared__ __align__(16) unsigned short ldsW[2][LDS_CHUNK_ELEMS];

    const int lane   = threadIdx.x & 31;
    const int wid    = (blockIdx.x << 3) | (threadIdx.x >> 5);  // 0..511
    const int m_tile = wid & 15;       // batch tile (16 rows)
    const int h_tile = wid >> 4;       // hidden tile; uniform per block

    const int ln16   = lane & 15;
    const int a_koff = (lane >> 4) << 4;   // 0 or 16 (A contiguous-K)
    const int b_koff = (lane >> 4) << 4;   // 0 or 16 (B contiguous-K)

    const int arow = m_tile * 16 + ln16;
    const unsigned short* xrow = x_t  + (size_t)arow * INDIM;
    const unsigned short* hrow = h_in + (size_t)arow * HDIM;

    // ---- weight-slice staging helper (chunk c -> ldsW[c&1]) ----
    auto stage_chunk = [&](int c) {
        const unsigned short* Wsrc = (c < 4) ? Wih : Whh;
        const int kbase = (c & 3) * KCHUNK;
        const size_t gelem = (size_t)h_tile * 16 * 512 + kbase;
#ifdef USE_TDM
        if (threadIdx.x < 32) {
            unsigned laddr = (unsigned)(size_t)(&ldsW[c & 1][0]);
            tdm_load_w_slice(laddr, Wsrc + gelem);
        }
#else
        // cooperative fallback: 1024 uint4 (16KB) across 256 threads
        const uint4* __restrict__ src4 = reinterpret_cast<const uint4*>(Wsrc);
        uint4* dst4 = reinterpret_cast<uint4*>(&ldsW[c & 1][0]);
        for (int idx = threadIdx.x; idx < LDS_CHUNK_ELEMS / 8; idx += 256) {
            int e    = idx * 8;                       // bf16 element offset
            int gate = e / (16 * KCHUNK);
            int rem  = e - gate * (16 * KCHUNK);
            int row  = rem / KCHUNK;
            int k    = rem - row * KCHUNK;
            size_t s = ((size_t)(gate * 512 + h_tile * 16 + row) * 512
                        + kbase + k) / 8;
            dst4[idx] = src4[s];
        }
#endif
    };

    // prologue: stage chunk 0
    stage_chunk(0);
#ifdef USE_TDM
    if (threadIdx.x < 32) __builtin_amdgcn_s_wait_tensorcnt(0);
#endif
    __syncthreads();

    v8f acc0 = {}, acc1 = {}, acc2 = {}, acc3 = {};

    for (int c = 0; c < NCHUNK; ++c) {
        // prefetch next chunk into the other buffer (safe: it was last read
        // in iteration c-1, whose trailing barrier has completed)
        if (c + 1 < NCHUNK) stage_chunk(c + 1);

        const unsigned short* Arow =
            (c < 4) ? (xrow + (c & 3) * KCHUNK) : (hrow + (c & 3) * KCHUNK);
        const unsigned short* Bl = &ldsW[c & 1][0];

#pragma unroll
        for (int kk = 0; kk < KCHUNK / 32; ++kk) {
            const int kb = kk * 32 + b_koff;
            v16bf a  = load_frag_a(Arow + kk * 32 + a_koff);
            v16bf b0 = load_frag_b(Bl + (0 * 16 + ln16) * KCHUNK + kb);
            v16bf b1 = load_frag_b(Bl + (1 * 16 + ln16) * KCHUNK + kb);
            v16bf b2 = load_frag_b(Bl + (2 * 16 + ln16) * KCHUNK + kb);
            v16bf b3 = load_frag_b(Bl + (3 * 16 + ln16) * KCHUNK + kb);
            acc0 = __builtin_amdgcn_wmma_f32_16x16x32_bf16(false, a, false, b0, (short)0, acc0, false, false);
            acc1 = __builtin_amdgcn_wmma_f32_16x16x32_bf16(false, a, false, b1, (short)0, acc1, false, false);
            acc2 = __builtin_amdgcn_wmma_f32_16x16x32_bf16(false, a, false, b2, (short)0, acc2, false, false);
            acc3 = __builtin_amdgcn_wmma_f32_16x16x32_bf16(false, a, false, b3, (short)0, acc3, false, false);
        }

#ifdef USE_TDM
        if (threadIdx.x < 32 && c + 1 < NCHUNK)
            __builtin_amdgcn_s_wait_tensorcnt(0);   // next chunk landed
#endif
        __syncthreads();   // publish next chunk / retire this buffer
    }

    // Per-lane bias scalars (C/D layout: N = lane%16, uniform over 8 rows)
    const int nb = h_tile * 16 + ln16;
    const float bi  = bias[0 * HDIM + nb];
    const float bf_ = bias[1 * HDIM + nb];
    const float bg  = bias[2 * HDIM + nb];
    const float bo_ = bias[3 * HDIM + nb];

    // ---- cell update + outputs ----
    const int j = h_tile * 16 + ln16;
    const int mbase = m_tile * 16 + ((lane >> 4) << 3);
#pragma unroll
    for (int r = 0; r < 8; ++r) {
        const int m = mbase + r;
        const size_t idx = (size_t)m * HDIM + j;

        const float cp = c_in[idx];
        const float iv = fast_sigmoid(acc0[r] + bi);
        const float fv = fast_sigmoid(acc1[r] + bf_);
        const float gv = tanhf(acc2[r] + bg);
        const float ov = fast_sigmoid(acc3[r] + bo_);
        const float cn = fv * cp + iv * gv;
        const float hn = ov * tanhf(cn);

        lstm_out_t[idx] = hn;

        if (has_next) {
            const float keep = 1.0f - (float)dones_next[m];
            const float hm = hn * keep;
            const float cm = cn * keep;
            hidden_out_next[idx] = hm;                          // pre-h (t+1)
            hidden_out_next[(size_t)BATCH * HDIM + idx] = cm;   // pre-c (t+1)
            h_out[idx] = f32_to_bf16_rne(hm);
            c_out[idx] = cm;
        }
    }
}

// ---------------------------------------------------------------------------
// Host side
// ---------------------------------------------------------------------------
extern "C" void kernel_launch(void* const* d_in, const int* in_sizes, int n_in,
                              void* d_out, int out_size, void* d_ws, size_t ws_size,
                              hipStream_t stream) {
    const float* x     = (const float*)d_in[0];   // [T,B,IN]
    const int*   dones = (const int*)d_in[1];     // [T,B]
    const float* W_ih  = (const float*)d_in[2];   // [4H,IN]
    const float* W_hh  = (const float*)d_in[3];   // [4H,H]
    const float* b_ih  = (const float*)d_in[4];   // [4H]
    const float* b_hh  = (const float*)d_in[5];   // [4H]

    float* out         = (float*)d_out;
    float* lstm_out    = out;                                   // [T,B,H]
    float* hidden_out  = out + (size_t)T_STEPS * BATCH * HDIM;  // [T,2,B,H]

    // workspace carve (256B aligned)
    char* ws = (char*)d_ws;
    size_t off = 0;
    auto take = [&](size_t bytes) {
        char* p = ws + off;
        off = (off + bytes + 255) & ~(size_t)255;
        return p;
    };
    unsigned short* xb  = (unsigned short*)take((size_t)T_STEPS * BATCH * INDIM * 2);
    unsigned short* Wib = (unsigned short*)take((size_t)G4 * INDIM * 2);
    unsigned short* Whb = (unsigned short*)take((size_t)G4 * HDIM * 2);
    float*          bia = (float*)take(G4 * 4);
    unsigned short* hA  = (unsigned short*)take((size_t)BATCH * HDIM * 2);
    unsigned short* hB  = (unsigned short*)take((size_t)BATCH * HDIM * 2);
    float*          cA  = (float*)take((size_t)BATCH * HDIM * 4);
    float*          cB  = (float*)take((size_t)BATCH * HDIM * 4);

    // prologue: convert to bf16, combine bias, zero state & hidden[0]
    cvt_f32_bf16_kernel<<<8192, 256, 0, stream>>>(x, xb, T_STEPS * BATCH * INDIM);
    cvt_f32_bf16_kernel<<<2048, 256, 0, stream>>>(W_ih, Wib, G4 * INDIM);
    cvt_f32_bf16_kernel<<<2048, 256, 0, stream>>>(W_hh, Whb, G4 * HDIM);
    init_kernel<<<(2 * BATCH * HDIM + 255) / 256, 256, 0, stream>>>(
        b_ih, b_hh, bia, hA, cA, hidden_out /* hidden_outputs[0] = zeros */);

    // sequential scan: one fused WMMA kernel per timestep, ping-pong state
    for (int t = 0; t < T_STEPS; ++t) {
        const int ping = t & 1;
        const unsigned short* h_in  = ping ? hB : hA;
        unsigned short*       h_out = ping ? hA : hB;
        const float*          c_in  = ping ? cB : cA;
        float*                c_out = ping ? cA : cB;
        const int has_next = (t + 1 < T_STEPS) ? 1 : 0;

        lstm_step_kernel<<<64, 256, 0, stream>>>(
            xb + (size_t)t * BATCH * INDIM,
            Wib, Whb, bia,
            h_in, c_in, h_out, c_out,
            dones + (size_t)(has_next ? (t + 1) : t) * BATCH,
            lstm_out + (size_t)t * BATCH * HDIM,
            hidden_out + (size_t)(t + 1) * 2 * BATCH * HDIM,
            has_next);
    }
}
